// FGHGNN_37941741093443
// MI455X (gfx1250) — compile-verified, matrix operand order checked by reading, not run on hardware
//
#include <hip/hip_runtime.h>
#include <hip/hip_bf16.h>
#include <math.h>

// ---------------------------------------------------------------------------
// FGHGNN forward for MI455X (gfx1250, wave32, WMMA).
//
// Sizes (from reference):
#define HID    128
#define PROJ   256
#define HEADS  4
#define OUTD   10
#define NATOM  80000
#define NCL    20000
#define NGRAPH 256
#define EATOM  640000
#define EA2C   80000
//
// Input leaf indexing assumption: top-level setup_inputs() dict in insertion
// order; 'params' flattened jax-style (dict keys sorted at every level,
// lists in order):
//   0 atom_emb  1 bond_emb  2 c2c_emb  3 cl_emb
//   4 cls_W1    5 cls_W2    6 cls_b1   7 cls_b2
//   8 + layer*44 : per-layer leaves, layer dict sorted
//     +0..7   atom2c_mlp   : stage{0,1} x (W,b,beta,gamma)
//     +8      conv0 eps, +9..16  conv0 mlp stage{0,1} x (W,b,beta,gamma)
//     +17     conv1 eps, +18..25 conv1 mlp
//     +26..33 c2atom_mlp   : stage{0,1} x (W,b,beta,gamma)
//     +34..38 pool   GAT   : W_dst, W_src, att_dst, att_src, bias
//     +39..43 unpool GAT   : W_dst, W_src, att_dst, att_src, bias
//   96 atom_ids  97 cluster_ids  98 bond_ids  99 c2c_ids
//   100 edge_index[2,E]  101 c2c_edge_index  102 atom2c_ei  103 c2atom_ei
//   104 x_batch  105 x_cluster_batch
//
// Workspace requirement: ~130M floats (~520 MB).
// ---------------------------------------------------------------------------

typedef __attribute__((ext_vector_type(16))) _Float16 v16h;
typedef __attribute__((ext_vector_type(8)))  _Float16 v8h;
typedef __attribute__((ext_vector_type(8)))  float    v8f;

// ---------------- generic GEMM: C[M,N] = A[M,K] @ B[K,N] (+bias) -----------
// f32 in/out, f16 WMMA compute with f32 accumulate. N % 64 == 0, K % 32 == 0.
//
// LDS tiles are stored pre-swizzled into the WMMA 16-bit fragment layout
// (CDNA5 ISA 7.12.2): K index c maps to (half = (c>>3)&1,
// slot = (c&16 ? 8:0) + (c&7)).  Each lane then loads its entire 16-element
// fragment as one contiguous 32-byte LDS read (2x ds_load_b128) instead of
// 16 scalar ds_load_u16 + v_mov_b16 packing.
__global__ void k_gemm(const float* __restrict__ A, const float* __restrict__ B,
                       const float* __restrict__ bias, float* __restrict__ C,
                       int M, int K, int N) {
  __shared__ _Float16 sA[128 * 32];  // [row][half*16 + slot]
  __shared__ _Float16 sB[32 * 64];   // [(t*2+half)*16 + n][slot]
  const int tid  = threadIdx.x;
  const int wid  = tid >> 5;         // 8 waves -> 8 row stripes of 16
  const int lane = tid & 31;
  const int half = lane >> 4;
  const int l16  = lane & 15;
  const long row0 = (long)blockIdx.x * 128;
  const int  col0 = blockIdx.y * 64;

  v8f acc[4] = {v8f{0}, v8f{0}, v8f{0}, v8f{0}};

  for (int k0 = 0; k0 < K; k0 += 32) {
    // ---- A tile: 128x32 = 512 K-octets; an octet (8 consecutive K) maps to
    // 8 contiguous swizzled slots -> one v8h (16B) LDS store per octet.
#pragma unroll
    for (int j = 0; j < 2; ++j) {
      int o  = tid + j * 256;        // octet id 0..511
      int r  = o >> 2;               // row 0..127
      int c0 = (o & 3) * 8;          // 0,8,16,24
      int ah = (c0 >> 3) & 1;
      int sl = (c0 & 16) ? 8 : 0;
      v8h hv;
      long gr = row0 + r;
      if (gr < M) {
        const float4* p = (const float4*)&A[gr * K + k0 + c0];
        float4 u = p[0], w = p[1];
        hv[0] = (_Float16)u.x; hv[1] = (_Float16)u.y;
        hv[2] = (_Float16)u.z; hv[3] = (_Float16)u.w;
        hv[4] = (_Float16)w.x; hv[5] = (_Float16)w.y;
        hv[6] = (_Float16)w.z; hv[7] = (_Float16)w.w;
        // WGP-scope prefetch of next K-tile (locality 3 -> near caches):
        // this block consumes it on the next iteration.
        if (k0 + 32 < K) __builtin_prefetch(&A[gr * K + k0 + 32 + c0], 0, 3);
      } else {
        hv = v8h{0};
      }
      *(v8h*)&sA[r * 32 + ah * 16 + sl] = hv;
    }
    // ---- B tile: 32x64 = 512 N-quads; coalesced float4 read along N,
    // fragment order runs along K so the 4 stores are stride-16 ds_store_u16.
#pragma unroll
    for (int j = 0; j < 2; ++j) {
      int q  = tid + j * 256;        // quad id 0..511
      int r  = q >> 4;               // K row 0..31
      int c0 = (q & 15) * 4;         // col 0..60
      float4 u = *(const float4*)&B[(long)(k0 + r) * N + col0 + c0];
      int bh = (r >> 3) & 1;
      int sl = ((r & 16) ? 8 : 0) + (r & 7);
      int t  = c0 >> 4;
      int base = ((t * 2 + bh) * 16 + (c0 & 15)) * 16 + sl;
      sB[base +  0] = (_Float16)u.x;
      sB[base + 16] = (_Float16)u.y;
      sB[base + 32] = (_Float16)u.z;
      sB[base + 48] = (_Float16)u.w;
    }
    __syncthreads();

    // Whole fragments as contiguous 32B LDS reads.
    v16h af = *(const v16h*)&sA[(wid * 16 + l16) * 32 + half * 16];
#pragma unroll
    for (int t = 0; t < 4; ++t) {
      v16h bf = *(const v16h*)&sB[((t * 2 + half) * 16 + l16) * 16];
      acc[t] = __builtin_amdgcn_wmma_f32_16x16x32_f16(
          false, af, false, bf, (short)0, acc[t], false, false);
    }
    __syncthreads();
  }

#pragma unroll
  for (int t = 0; t < 4; ++t) {
    int gn = col0 + t * 16 + l16;
    float bs = bias ? bias[gn] : 0.f;
#pragma unroll
    for (int r = 0; r < 8; ++r) {
      long gm = row0 + wid * 16 + r + 8 * half;  // C/D VGPR r -> row r + 8*half
      if (gm < M) C[gm * N + gn] = acc[t][r] + bs;
    }
  }
}

// ---------------- elementwise / scatter helpers ----------------------------
__global__ void k_fill(float* p, float v, long n) {
  long i = (long)blockIdx.x * 256 + threadIdx.x;
  if (i < n) p[i] = v;
}

__global__ void k_copy(float* dst, const float* src, long n) {
  long i = (long)blockIdx.x * 256 + threadIdx.x;
  if (i < n) dst[i] = src[i];
}

__global__ void k_relu(float* p, long n) {
  long i = (long)blockIdx.x * 256 + threadIdx.x;
  if (i < n) p[i] = p[i] > 0.f ? p[i] : 0.f;
}

__global__ void k_gather(const float* __restrict__ emb, const int* __restrict__ ids,
                         float* __restrict__ out, long n, int dim) {
  long i = (long)blockIdx.x * 256 + threadIdx.x;
  if (i >= n) return;
  long node = i / dim; int c = (int)(i % dim);
  out[i] = emb[(long)ids[node] * dim + c];
}

// GINE: agg[dst] += relu(x[src] + bond_emb[bond_id]); bond table stays in L2.
__global__ void k_gine_scatter(const float* __restrict__ x, const float* __restrict__ bemb,
                               const int* __restrict__ bids, const int* __restrict__ src,
                               const int* __restrict__ dst, float* __restrict__ agg, int E) {
  long i = (long)blockIdx.x * 256 + threadIdx.x;
  if (i >= (long)E * HID) return;
  int e = (int)(i >> 7), f = (int)(i & 127);
  float m = x[(long)src[e] * HID + f] + bemb[(long)bids[e] * HID + f];
  m = m > 0.f ? m : 0.f;
  atomicAdd(&agg[(long)dst[e] * HID + f], m);
}

__global__ void k_gine_combine(const float* __restrict__ x, const float* __restrict__ agg,
                               const float* __restrict__ epsp, float* __restrict__ z, long n) {
  long i = (long)blockIdx.x * 256 + threadIdx.x;
  if (i >= n) return;
  z[i] = (1.f + *epsp) * x[i] + agg[i];
}

// Column sum / sum-of-squares for training-mode BatchNorm.
__global__ void k_colstats(const float* __restrict__ Y, float* __restrict__ s,
                           float* __restrict__ q, int M, int N, int rows_per_blk) {
  int col = blockIdx.x * 64 + (threadIdx.x & 63);
  int rg  = threadIdx.x >> 6;  // 0..3
  int r0  = blockIdx.y * rows_per_blk;
  int r1  = min(M, r0 + rows_per_blk);
  float ps = 0.f, pq = 0.f;
  for (int r = r0 + rg; r < r1; r += 4) {
    float v = Y[(long)r * N + col];
    ps += v; pq += v * v;
  }
  atomicAdd(&s[col], ps);
  atomicAdd(&q[col], pq);
}

__global__ void k_bn_relu(const float* __restrict__ Y, const float* __restrict__ s,
                          const float* __restrict__ q, const float* __restrict__ gamma,
                          const float* __restrict__ beta, const float* __restrict__ res,
                          float* __restrict__ out, int M, int N) {
  long i = (long)blockIdx.x * 256 + threadIdx.x;
  if (i >= (long)M * N) return;
  int c = (int)(i % N);
  float mu  = s[c] / (float)M;
  float var = q[c] / (float)M - mu * mu;
  float v = gamma[c] * (Y[i] - mu) * rsqrtf(var + 1e-5f) + beta[c];
  v = v > 0.f ? v : 0.f;
  if (res) v += res[i];
  out[i] = v;
}

// ---------------- GAT kernels ----------------------------------------------
// Vd[i,h] = sum_k W_dst[i, h*128+k] * att_dst[h,k]   (128x4 folded matrix)
__global__ void k_vdst(const float* __restrict__ Wd, const float* __restrict__ attd,
                       float* __restrict__ Vd) {
  int t = threadIdx.x;                 // 0..511
  int i = t >> 2, h = t & 3;
  float sum = 0.f;
  for (int k = 0; k < HID; ++k)
    sum += Wd[(long)i * (HEADS * HID) + h * HID + k] * attd[h * HID + k];
  Vd[i * 4 + h] = sum;
}

__global__ void k_alpha_d(const float* __restrict__ xd, const float* __restrict__ Vd,
                          float* __restrict__ ad, int Nd) {
  long i = (long)blockIdx.x * 256 + threadIdx.x;
  if (i >= (long)Nd * 4) return;
  int n = (int)(i >> 2), h = (int)(i & 3);
  float sum = 0.f;
  for (int k = 0; k < HID; ++k) sum += xd[(long)n * HID + k] * Vd[k * 4 + h];
  ad[i] = sum;
}

__global__ void k_alpha_s(const float* __restrict__ hs, const float* __restrict__ atts,
                          float* __restrict__ as_, int Ns) {
  long i = (long)blockIdx.x * 256 + threadIdx.x;
  if (i >= (long)Ns * 4) return;
  int n = (int)(i >> 2), h = (int)(i & 3);
  float sum = 0.f;
  for (int k = 0; k < HID; ++k)
    sum += hs[(long)n * (HEADS * HID) + h * HID + k] * atts[h * HID + k];
  as_[i] = sum;
}

static __device__ __forceinline__ void atomicMaxF(float* addr, float v) {
  if (v >= 0.f) atomicMax((int*)addr, __float_as_int(v));
  else          atomicMin((unsigned int*)addr, __float_as_uint(v));
}

__global__ void k_edge_max(const int* __restrict__ src, const int* __restrict__ dst,
                           const float* __restrict__ as_, const float* __restrict__ ad,
                           float* __restrict__ araw, float* __restrict__ mmax, int E) {
  long i = (long)blockIdx.x * 256 + threadIdx.x;
  if (i >= (long)E * 4) return;
  int e = (int)(i >> 2), h = (int)(i & 3);
  float a = as_[(long)src[e] * 4 + h] + ad[(long)dst[e] * 4 + h];
  a = a > 0.f ? a : 0.2f * a;  // leaky_relu 0.2
  araw[i] = a;
  atomicMaxF(&mmax[(long)dst[e] * 4 + h], a);
}

__global__ void k_edge_exp(const int* __restrict__ dst, float* __restrict__ araw,
                           const float* __restrict__ mmax, float* __restrict__ den, int E) {
  long i = (long)blockIdx.x * 256 + threadIdx.x;
  if (i >= (long)E * 4) return;
  int e = (int)(i >> 2), h = (int)(i & 3);
  float p = __expf(araw[i] - mmax[(long)dst[e] * 4 + h]);
  araw[i] = p;
  atomicAdd(&den[(long)dst[e] * 4 + h], p);
}

__global__ void k_bias_rows(float* __restrict__ out, const float* __restrict__ bias, int Nd) {
  long i = (long)blockIdx.x * 256 + threadIdx.x;
  if (i >= (long)Nd * (HEADS * HID)) return;
  out[i] = bias[(int)(i % (HEADS * HID))];
}

__global__ void k_edge_scatter(const int* __restrict__ src, const int* __restrict__ dst,
                               const float* __restrict__ araw, const float* __restrict__ den,
                               const float* __restrict__ hs, float* __restrict__ out, int E) {
  long i = (long)blockIdx.x * 256 + threadIdx.x;
  if (i >= (long)E * HID) return;
  int e = (int)(i >> 7), f = (int)(i & 127);
  long s = src[e], d = dst[e];
#pragma unroll
  for (int h = 0; h < HEADS; ++h) {
    float c = araw[(long)e * 4 + h] / (den[d * 4 + h] + 1e-16f);
    atomicAdd(&out[d * (HEADS * HID) + h * HID + f],
              c * hs[s * (HEADS * HID) + h * HID + f]);
  }
}

// ---------------- readout ---------------------------------------------------
__global__ void k_pool_sum(const float* __restrict__ x, const int* __restrict__ batch,
                           float* __restrict__ psum, float* __restrict__ cnt, int Nn) {
  long i = (long)blockIdx.x * 256 + threadIdx.x;
  if (i >= (long)Nn * HID) return;
  int n = (int)(i >> 7), f = (int)(i & 127);
  int b = batch[n];
  atomicAdd(&psum[(long)b * HID + f], x[i]);
  if (f == 0) atomicAdd(&cnt[b], 1.f);
}

__global__ void k_pool_final(const float* __restrict__ pa, const float* __restrict__ ca,
                             const float* __restrict__ pc, const float* __restrict__ cc,
                             float* __restrict__ pooled) {
  long i = (long)blockIdx.x * 256 + threadIdx.x;
  if (i >= (long)NGRAPH * 2 * HID) return;
  int g = (int)(i / (2 * HID)), c = (int)(i % (2 * HID));
  float v;
  if (c < HID) v = pa[(long)g * HID + c] / fmaxf(ca[g], 1.f);
  else         v = pc[(long)g * HID + (c - HID)] / fmaxf(cc[g], 1.f);
  pooled[i] = v;
}

__global__ void k_cls2(const float* __restrict__ h, const float* __restrict__ W,
                       const float* __restrict__ b, float* __restrict__ out) {
  int i = blockIdx.x * 256 + threadIdx.x;
  if (i >= NGRAPH * OUTD) return;
  int g = i / OUTD, o = i % OUTD;
  float s = b[o];
  for (int k = 0; k < 2 * HID; ++k) s += h[(long)g * 2 * HID + k] * W[(long)k * OUTD + o];
  out[i] = s;
}

// ---------------------------------------------------------------------------
extern "C" void kernel_launch(void* const* d_in, const int* in_sizes, int n_in,
                              void* d_out, int out_size, void* d_ws, size_t ws_size,
                              hipStream_t stream) {
  (void)in_sizes; (void)n_in; (void)out_size; (void)ws_size;
  auto F = [&](int i) { return (const float*)d_in[i]; };
  auto I = [&](int i) { return (const int*)d_in[i]; };
  auto blocks = [](long n) { return dim3((unsigned)((n + 255) / 256)); };

  // ---- workspace bump allocator (floats); peak ~130M floats (~520 MB) ----
  float* ws = (float*)d_ws;
  long o = 0;
  auto alloc = [&](long n) { float* p = ws + o; o += n; return p; };
  float* X    = alloc((long)NATOM * HID);
  float* XCL  = alloc((long)NCL * HID);
  float* H    = alloc((long)NATOM * HID);
  float* HCL  = alloc((long)NCL * HID);
  float* BIG1 = alloc((long)NATOM * HEADS * HID);   // gat out / agg+y2
  float* BIG2 = alloc((long)NATOM * HEADS * HID);   // hs buffers
  float* MID  = alloc((long)NATOM * PROJ);          // MLP hidden
  float* AS   = alloc((long)NATOM * 4);
  float* AD   = alloc((long)NATOM * 4);
  float* ARAW = alloc((long)EA2C * 4);
  float* MMAX = alloc((long)NATOM * 4);
  float* DEN  = alloc((long)NATOM * 4);
  float* VD   = alloc(512);
  float* CSUM = alloc(512);
  float* CSQ  = alloc(512);
  float* PSA  = alloc((long)NGRAPH * HID);
  float* PSC  = alloc((long)NGRAPH * HID);
  float* CTA  = alloc(NGRAPH);
  float* CTC  = alloc(NGRAPH);
  float* POOL = alloc((long)NGRAPH * 2 * HID);
  float* CLS1 = alloc((long)NGRAPH * PROJ);

  const int* atom_ids = I(96);
  const int* cl_ids   = I(97);
  const int* bond_ids = I(98);
  const int* ei_src   = I(100);
  const int* ei_dst   = I(100) + EATOM;
  const int* a2c_src  = I(102);
  const int* a2c_dst  = I(102) + EA2C;
  const int* c2a_src  = I(103);
  const int* c2a_dst  = I(103) + EA2C;

  auto fill = [&](float* p, float v, long n) {
    k_fill<<<blocks(n), 256, 0, stream>>>(p, v, n);
  };
  auto gemm = [&](const float* A, const float* B, const float* bias, float* C,
                  int M, int K, int N) {
    dim3 g((unsigned)((M + 127) / 128), (unsigned)(N / 64));
    k_gemm<<<g, 256, 0, stream>>>(A, B, bias, C, M, K, N);
  };
  // MLP stage: Linear -> BN(batch stats) -> ReLU (+optional residual), leaf0 = W.
  auto mlp_stage = [&](const float* A, int M, int K, int N, int leaf0,
                       const float* res, float* Y, float* out) {
    gemm(A, F(leaf0), F(leaf0 + 1), Y, M, K, N);
    fill(CSUM, 0.f, N); fill(CSQ, 0.f, N);
    dim3 gs((unsigned)(N / 64), (unsigned)((M + 4095) / 4096));
    k_colstats<<<gs, 256, 0, stream>>>(Y, CSUM, CSQ, M, N, 4096);
    k_bn_relu<<<blocks((long)M * N), 256, 0, stream>>>(
        Y, CSUM, CSQ, F(leaf0 + 3) /*gamma*/, F(leaf0 + 2) /*beta*/, res, out, M, N);
  };
  // Bipartite multi-head GAT; gleaf = W_dst leaf.
  auto gat = [&](int gleaf, const float* xs, int Ns, const float* xd, int Nd,
                 const int* esrc, const int* edst, int E, float* hs, float* outb) {
    gemm(xs, F(gleaf + 1) /*W_src*/, nullptr, hs, Ns, HID, HEADS * HID);
    k_alpha_s<<<blocks((long)Ns * 4), 256, 0, stream>>>(hs, F(gleaf + 3), AS, Ns);
    k_vdst<<<1, 512, 0, stream>>>(F(gleaf), F(gleaf + 2), VD);
    k_alpha_d<<<blocks((long)Nd * 4), 256, 0, stream>>>(xd, VD, AD, Nd);
    fill(MMAX, -INFINITY, (long)Nd * 4);
    fill(DEN, 0.f, (long)Nd * 4);
    k_edge_max<<<blocks((long)E * 4), 256, 0, stream>>>(esrc, edst, AS, AD, ARAW, MMAX, E);
    k_edge_exp<<<blocks((long)E * 4), 256, 0, stream>>>(edst, ARAW, MMAX, DEN, E);
    k_bias_rows<<<blocks((long)Nd * HEADS * HID), 256, 0, stream>>>(outb, F(gleaf + 4), Nd);
    k_edge_scatter<<<blocks((long)E * HID), 256, 0, stream>>>(esrc, edst, ARAW, DEN, hs, outb, E);
  };
  // GINE conv with residual, in-place on X. convleaf = eps leaf.
  auto gine = [&](int convleaf) {
    float* AGG = BIG1;
    float* Y2  = BIG1 + (long)NATOM * HID;
    fill(AGG, 0.f, (long)NATOM * HID);
    k_gine_scatter<<<blocks((long)EATOM * HID), 256, 0, stream>>>(
        X, F(1) /*bond_emb*/, bond_ids, ei_src, ei_dst, AGG, EATOM);
    k_gine_combine<<<blocks((long)NATOM * HID), 256, 0, stream>>>(
        X, AGG, F(convleaf), AGG, (long)NATOM * HID);
    mlp_stage(AGG, NATOM, HID, PROJ, convleaf + 1, nullptr, MID, MID);
    mlp_stage(MID, NATOM, PROJ, HID, convleaf + 5, X /*residual*/, Y2, X);
  };

  // ---- forward ----
  k_gather<<<blocks((long)NATOM * HID), 256, 0, stream>>>(F(0), atom_ids, X, (long)NATOM * HID, HID);
  k_gather<<<blocks((long)NCL * HID), 256, 0, stream>>>(F(3), cl_ids, XCL, (long)NCL * HID, HID);

  for (int L = 0; L < 2; ++L) {
    int lb = 8 + L * 44;
    gine(lb + 8);    // conv0
    gine(lb + 17);   // conv1
    // h = c2atom_mlp(GAT(unpool, xs=x_cl, xd=x)) + x
    gat(lb + 39, XCL, NCL, X, NATOM, c2a_src, c2a_dst, EA2C, BIG2, BIG1);
    mlp_stage(BIG1, NATOM, HEADS * HID, PROJ, lb + 26, nullptr, MID, MID);
    mlp_stage(MID, NATOM, PROJ, HID, lb + 30, X, H, H);
    // h_cl = atom2c_mlp(GAT(pool, xs=x, xd=x_cl)) + x_cl
    gat(lb + 34, X, NATOM, XCL, NCL, a2c_src, a2c_dst, EA2C, BIG2, BIG1);
    mlp_stage(BIG1, NCL, HEADS * HID, PROJ, lb + 0, nullptr, MID, MID);
    mlp_stage(MID, NCL, PROJ, HID, lb + 4, XCL, HCL, HCL);
    k_copy<<<blocks((long)NATOM * HID), 256, 0, stream>>>(X, H, (long)NATOM * HID);
    k_copy<<<blocks((long)NCL * HID), 256, 0, stream>>>(XCL, HCL, (long)NCL * HID);
  }

  // readout: per-graph mean pool, concat, 2-layer classifier
  fill(PSA, 0.f, (long)NGRAPH * HID); fill(PSC, 0.f, (long)NGRAPH * HID);
  fill(CTA, 0.f, NGRAPH);             fill(CTC, 0.f, NGRAPH);
  k_pool_sum<<<blocks((long)NATOM * HID), 256, 0, stream>>>(X, I(104), PSA, CTA, NATOM);
  k_pool_sum<<<blocks((long)NCL * HID), 256, 0, stream>>>(XCL, I(105), PSC, CTC, NCL);
  k_pool_final<<<blocks((long)NGRAPH * 2 * HID), 256, 0, stream>>>(PSA, CTA, PSC, CTC, POOL);
  gemm(POOL, F(4) /*cls_W1*/, F(6) /*cls_b1*/, CLS1, NGRAPH, 2 * HID, PROJ);
  k_relu<<<blocks((long)NGRAPH * PROJ), 256, 0, stream>>>(CLS1, (long)NGRAPH * PROJ);
  k_cls2<<<blocks((long)NGRAPH * OUTD), 256, 0, stream>>>(CLS1, F(5), F(7), (float*)d_out);
}
